// MHSA_42030549958927
// MI455X (gfx1250) — matrix operand order repaired and checked by reading, not compiled, not run on hardware
//
#include <hip/hip_runtime.h>
#include <hip/hip_bf16.h>
#include <math.h>

typedef __attribute__((ext_vector_type(16))) _Float16 v16h;
typedef __attribute__((ext_vector_type(8)))  _Float16 v8h;
typedef __attribute__((ext_vector_type(8)))  float    v8f;

#define DM    1024
#define NH    16
#define DH    64
#define BS    2
#define SEQ   2048
#define MTOT  (BS * SEQ)   // 4096 token rows

// ---------------------------------------------------------------------------
// WMMA fragment loaders (wave32, v_wmma_f32_16x16x32_f16).
//
// A (16x32 f16): lane L holds row m=L&15, K run {kb..kb+15},
// kb = (L>=16)?16:0 (same order as B).  Two 16B loads.
//
// B (32x16 f16): lane L holds column n=L&15, K run {kb..kb+15}, kb=(L>=16)?16:0.
// With B[k][n] = W[n][k] (W row-major (N,K)), that is one contiguous 32B row
// segment of W -> two 16B loads, no transpose needed.
// ---------------------------------------------------------------------------
static __device__ __forceinline__ v16h load_a_f16(const _Float16* A, int ld,
                                                  int row0, int k0) {
  const int lane = threadIdx.x & 31;
  const int m    = lane & 15;
  const int kb   = (lane >> 4) << 4;             // 0 or 16
  const _Float16* p = A + (size_t)(row0 + m) * ld + k0 + kb;
  v8h lo = *(const v8h*)(p);
  v8h hi = *(const v8h*)(p + 8);
  v16h r;
#pragma unroll
  for (int i = 0; i < 8; ++i) { r[i] = lo[i]; r[i + 8] = hi[i]; }
  return r;
}

static __device__ __forceinline__ v16h load_b_f16(const _Float16* W, int ld,
                                                  int n0, int k0) {
  const int lane = threadIdx.x & 31;
  const int n    = lane & 15;
  const int kb   = (lane >> 4) << 4;             // 0 or 16
  const _Float16* p = W + (size_t)(n0 + n) * ld + k0 + kb;
  v8h lo = *(const v8h*)(p);
  v8h hi = *(const v8h*)(p + 8);
  v16h r;
#pragma unroll
  for (int i = 0; i < 8; ++i) { r[i] = lo[i]; r[i + 8] = hi[i]; }
  return r;
}

static __device__ __forceinline__ v8f wmma_f16(v16h a, v16h b, v8f c) {
  return __builtin_amdgcn_wmma_f32_16x16x32_f16(
      /*neg_a=*/false, a, /*neg_b=*/false, b,
      /*c_mod=*/(short)0, c, /*reuse_a=*/false, /*reuse_b=*/false);
}

// ---------------------------------------------------------------------------
// Kernel 0: f32 -> f16 conversion (inputs are tiny vs 23.3 TB/s)
// ---------------------------------------------------------------------------
__global__ void f32_to_f16_kern(const float* __restrict__ in,
                                _Float16* __restrict__ out, int n) {
  int i = blockIdx.x * blockDim.x + threadIdx.x;
  if (i < n) out[i] = (_Float16)in[i];
}

// ---------------------------------------------------------------------------
// Kernel 1: QKV GEMM (xh[4096,1024] x wqkv[3072,1024]^T) with fused RoPE.
// 64x64 tile / block, 4 waves, each wave a 32x32 quadrant (2x2 WMMA frags).
// Epilogue: stage f32 tile in LDS, rotate interleaved pairs, scatter f16 to
// Q,K in (B,H,S,D) and V transposed to (B,H,D,S).
// ---------------------------------------------------------------------------
__global__ __launch_bounds__(128)
void qkv_rope_gemm(const _Float16* __restrict__ xh,
                   const _Float16* __restrict__ wh,
                   const int* __restrict__ pos,
                   _Float16* __restrict__ Qh, _Float16* __restrict__ Kh,
                   _Float16* __restrict__ Vt) {
  __shared__ float tile[64 * 66];
  const int tm   = blockIdx.x * 64;
  const int tn   = blockIdx.y * 64;
  const int wave = threadIdx.x >> 5;
  const int lane = threadIdx.x & 31;
  const int wr   = (wave >> 1) * 32;
  const int wc   = (wave & 1) * 32;
  const int mh   = (lane >> 4) << 3;  // +8 row offset for upper half-wave
  const int nl   = lane & 15;

  v8f c00 = {}, c01 = {}, c10 = {}, c11 = {};
  for (int k0 = 0; k0 < DM; k0 += 32) {
    v16h a0 = load_a_f16(xh, DM, tm + wr,      k0);
    v16h a1 = load_a_f16(xh, DM, tm + wr + 16, k0);
    v16h b0 = load_b_f16(wh, DM, tn + wc,      k0);
    v16h b1 = load_b_f16(wh, DM, tn + wc + 16, k0);
    c00 = wmma_f16(a0, b0, c00);
    c01 = wmma_f16(a0, b1, c01);
    c10 = wmma_f16(a1, b0, c10);
    c11 = wmma_f16(a1, b1, c11);
  }

#pragma unroll
  for (int v = 0; v < 8; ++v) {
    tile[(wr +      mh + v) * 66 + (wc +      nl)] = c00[v];
    tile[(wr +      mh + v) * 66 + (wc + 16 + nl)] = c01[v];
    tile[(wr + 16 + mh + v) * 66 + (wc +      nl)] = c10[v];
    tile[(wr + 16 + mh + v) * 66 + (wc + 16 + nl)] = c11[v];
  }
  __syncthreads();

  const int which = tn >> 10;          // 0=q 1=k 2=v (64-col tile == one head)
  const int h     = (tn & 1023) >> 6;
  for (int p = threadIdx.x; p < 64 * 32; p += 128) {
    const int row = p >> 5;
    const int i   = p & 31;            // pair index: dims (2i, 2i+1)
    const int t   = tm + row;
    const int b   = t >> 11;           // S = 2048
    const int s   = t & (SEQ - 1);
    const float xr = tile[row * 66 + 2 * i];
    const float xi = tile[row * 66 + 2 * i + 1];
    if (which < 2) {
      const float pf  = (float)pos[s];
      const float inv = __powf(10000.0f, -(2.0f * (float)i) / 64.0f);
      float sn, cs;
      __sincosf(pf * inv, &sn, &cs);
      const float orr = xr * cs - xi * sn;
      const float oi  = xr * sn + xi * cs;
      _Float16* dst = (which == 0) ? Qh : Kh;
      const size_t base = ((size_t)(b * NH + h) * SEQ + s) * DH;
      dst[base + 2 * i]     = (_Float16)orr;
      dst[base + 2 * i + 1] = (_Float16)oi;
    } else {
      const size_t base = (size_t)(b * NH + h) * DH * SEQ;   // (B,H,D,S)
      Vt[base + (size_t)(2 * i)     * SEQ + s] = (_Float16)xr;
      Vt[base + (size_t)(2 * i + 1) * SEQ + s] = (_Float16)xi;
    }
  }
}

// ---------------------------------------------------------------------------
// Kernel 2: causal flash attention. grid = (S/64, B*H), 4 waves per block,
// each wave owns 16 query rows. Per 32-key chunk: 4 WMMAs for Q.K^T,
// streaming softmax (shfl_xor row reductions), P staged through per-wave LDS
// (C-layout -> A-layout transpose), 4 WMMAs for P.V.
// ---------------------------------------------------------------------------
__global__ __launch_bounds__(128)
void flash_attn(const _Float16* __restrict__ Qh, const _Float16* __restrict__ Kh,
                const _Float16* __restrict__ Vt, _Float16* __restrict__ Oh) {
  __shared__ _Float16 pbuf[4][16][32];
  const int qt   = blockIdx.x;
  const int bh   = blockIdx.y;
  const int b    = bh >> 4;
  const int h    = bh & 15;
  const int wave = threadIdx.x >> 5;
  const int lane = threadIdx.x & 31;
  const int mh   = (lane >> 4) << 3;
  const int nl   = lane & 15;

  const _Float16* Qbh = Qh + (size_t)bh * SEQ * DH;
  const _Float16* Kbh = Kh + (size_t)bh * SEQ * DH;
  const _Float16* Vbh = Vt + (size_t)bh * DH * SEQ;

  const int qrow0 = qt * 64 + wave * 16;
  const v16h aq0 = load_a_f16(Qbh, DH, qrow0, 0);
  const v16h aq1 = load_a_f16(Qbh, DH, qrow0, 32);

  v8f o0 = {}, o1 = {}, o2 = {}, o3 = {};
  float mrow[8], lrow[8];
#pragma unroll
  for (int v = 0; v < 8; ++v) { mrow[v] = -1e30f; lrow[v] = 0.0f; }

  const float scale   = 0.125f;          // 1/sqrt(64)
  const int   nchunks = qt * 2 + 2;      // uniform per block: keys <= qt*64+63
  for (int c = 0; c < nchunks; ++c) {
    const int j = c * 32;
    v8f s0 = {}, s1 = {};
    s0 = wmma_f16(aq0, load_b_f16(Kbh, DH, j,       0),  s0);
    s0 = wmma_f16(aq1, load_b_f16(Kbh, DH, j,      32),  s0);
    s1 = wmma_f16(aq0, load_b_f16(Kbh, DH, j + 16,  0),  s1);
    s1 = wmma_f16(aq1, load_b_f16(Kbh, DH, j + 16, 32),  s1);

#pragma unroll
    for (int v = 0; v < 8; ++v) {
      const int q = qrow0 + mh + v;
      float e0 = s0[v] * scale; if (j + nl > q)      e0 = -1e30f;
      float e1 = s1[v] * scale; if (j + 16 + nl > q) e1 = -1e30f;

      float cmax = fmaxf(e0, e1);
#pragma unroll
      for (int off = 8; off >= 1; off >>= 1)
        cmax = fmaxf(cmax, __shfl_xor(cmax, off, 32));
      const float mnew  = fmaxf(mrow[v], cmax);
      const float alpha = __expf(mrow[v] - mnew);
      const float p0    = __expf(e0 - mnew);
      const float p1    = __expf(e1 - mnew);
      float ls = p0 + p1;
#pragma unroll
      for (int off = 8; off >= 1; off >>= 1)
        ls += __shfl_xor(ls, off, 32);
      lrow[v] = lrow[v] * alpha + ls;
      mrow[v] = mnew;
      o0[v] *= alpha; o1[v] *= alpha; o2[v] *= alpha; o3[v] *= alpha;

      pbuf[wave][mh + v][nl]      = (_Float16)p0;
      pbuf[wave][mh + v][16 + nl] = (_Float16)p1;
    }
    // intra-wave DS RAW: C-layout stores -> A-layout loads of the same tile
    asm volatile("s_wait_dscnt 0" ::: "memory");
    const v16h pa = load_a_f16(&pbuf[wave][0][0], 32, 0, 0);
    o0 = wmma_f16(pa, load_b_f16(Vbh, SEQ, 0,  j), o0);
    o1 = wmma_f16(pa, load_b_f16(Vbh, SEQ, 16, j), o1);
    o2 = wmma_f16(pa, load_b_f16(Vbh, SEQ, 32, j), o2);
    o3 = wmma_f16(pa, load_b_f16(Vbh, SEQ, 48, j), o3);
  }

#pragma unroll
  for (int v = 0; v < 8; ++v) {
    const float rl = 1.0f / lrow[v];
    const int   s  = qrow0 + mh + v;
    const size_t base = ((size_t)b * SEQ + s) * DM + h * DH;  // (B,S,H*D)
    Oh[base + 0  + nl] = (_Float16)(o0[v] * rl);
    Oh[base + 16 + nl] = (_Float16)(o1[v] * rl);
    Oh[base + 32 + nl] = (_Float16)(o2[v] * rl);
    Oh[base + 48 + nl] = (_Float16)(o3[v] * rl);
  }
}

// ---------------------------------------------------------------------------
// Kernel 3: output projection, Oh[4096,1024] x wout[1024,1024]^T -> f32 out
// ---------------------------------------------------------------------------
__global__ __launch_bounds__(128)
void out_proj(const _Float16* __restrict__ Oh, const _Float16* __restrict__ wh,
              float* __restrict__ out) {
  const int tm   = blockIdx.x * 64;
  const int tn   = blockIdx.y * 64;
  const int wave = threadIdx.x >> 5;
  const int lane = threadIdx.x & 31;
  const int wr   = (wave >> 1) * 32;
  const int wc   = (wave & 1) * 32;
  const int mh   = (lane >> 4) << 3;
  const int nl   = lane & 15;

  v8f c00 = {}, c01 = {}, c10 = {}, c11 = {};
  for (int k0 = 0; k0 < DM; k0 += 32) {
    v16h a0 = load_a_f16(Oh, DM, tm + wr,      k0);
    v16h a1 = load_a_f16(Oh, DM, tm + wr + 16, k0);
    v16h b0 = load_b_f16(wh, DM, tn + wc,      k0);
    v16h b1 = load_b_f16(wh, DM, tn + wc + 16, k0);
    c00 = wmma_f16(a0, b0, c00);
    c01 = wmma_f16(a0, b1, c01);
    c10 = wmma_f16(a1, b0, c10);
    c11 = wmma_f16(a1, b1, c11);
  }
#pragma unroll
  for (int v = 0; v < 8; ++v) {
    out[(size_t)(tm + wr +      mh + v) * DM + tn + wc +      nl] = c00[v];
    out[(size_t)(tm + wr +      mh + v) * DM + tn + wc + 16 + nl] = c01[v];
    out[(size_t)(tm + wr + 16 + mh + v) * DM + tn + wc +      nl] = c10[v];
    out[(size_t)(tm + wr + 16 + mh + v) * DM + tn + wc + 16 + nl] = c11[v];
  }
}

// ---------------------------------------------------------------------------
extern "C" void kernel_launch(void* const* d_in, const int* in_sizes, int n_in,
                              void* d_out, int out_size, void* d_ws,
                              size_t ws_size, hipStream_t stream) {
  const float* x    = (const float*)d_in[0];
  const int*   pos  = (const int*)d_in[1];
  const float* wqkv = (const float*)d_in[2];
  const float* wout = (const float*)d_in[3];
  float*       out  = (float*)d_out;

  // workspace carve-up (all f16): 8+6+2+8+8+8+8 = 48 MB
  char* ws = (char*)d_ws;
  _Float16* xh  = (_Float16*)ws; ws += (size_t)MTOT * DM * 2;
  _Float16* wqh = (_Float16*)ws; ws += (size_t)3 * DM * DM * 2;
  _Float16* woh = (_Float16*)ws; ws += (size_t)DM * DM * 2;
  _Float16* Qh  = (_Float16*)ws; ws += (size_t)BS * NH * SEQ * DH * 2;
  _Float16* Kh  = (_Float16*)ws; ws += (size_t)BS * NH * SEQ * DH * 2;
  _Float16* Vt  = (_Float16*)ws; ws += (size_t)BS * NH * SEQ * DH * 2;
  _Float16* Oh  = (_Float16*)ws; ws += (size_t)MTOT * DM * 2;

  int n;
  n = MTOT * DM;
  f32_to_f16_kern<<<(n + 255) / 256, 256, 0, stream>>>(x, xh, n);
  n = 3 * DM * DM;
  f32_to_f16_kern<<<(n + 255) / 256, 256, 0, stream>>>(wqkv, wqh, n);
  n = DM * DM;
  f32_to_f16_kern<<<(n + 255) / 256, 256, 0, stream>>>(wout, woh, n);

  qkv_rope_gemm<<<dim3(MTOT / 64, 3 * DM / 64), 128, 0, stream>>>(
      xh, wqh, pos, Qh, Kh, Vt);
  flash_attn<<<dim3(SEQ / 64, BS * NH), 128, 0, stream>>>(Qh, Kh, Vt, Oh);
  out_proj<<<dim3(MTOT / 64, DM / 64), 128, 0, stream>>>(Oh, woh, out);
}
